// PcLinearBlock_90245852823594
// MI455X (gfx1250) — matrix operand order repaired and checked by reading, not compiled, not run on hardware
//
#include <hip/hip_runtime.h>
#include <hip/hip_bf16.h>

// ---------------------------------------------------------------------------
// Predictive-coding relaxation on MI455X (gfx1250), bf16 WMMA GEMMs.
// ARCH = (1024, 2048, 2048, 512), BATCH = 4096, STEPS = 20, LR = 0.1
// ---------------------------------------------------------------------------

typedef __attribute__((ext_vector_type(16))) __bf16 v16bf;
typedef __attribute__((ext_vector_type(8)))  float  v8f;
typedef __attribute__((ext_vector_type(4)))  unsigned int u32x4;

union FragAB {
    u32x4 q[2];   // two 16-byte chunks = 8 VGPRs
    v16bf v;      // 16 bf16 per lane
};

__device__ __forceinline__ float bf2f(__bf16 x) { return (float)x; }
__device__ __forceinline__ __bf16 f2bf(float x) { return (__bf16)x; }

__device__ __forceinline__ float fast_tanh(float x) {
    // tanh(x) = 1 - 2/(e^{2x}+1); robust at +-inf, uses v_exp_f32.
    float e = __expf(2.0f * x);
    return 1.0f - 2.0f / (e + 1.0f);
}

// ---------------------------------------------------------------------------
// Fused GEMM: acc = A(MxK, bf16 row-major) x B(KxN), where B is supplied as
// BT (NxK row-major, bf16).  M is given by grid (blockIdx.y * 128).
//
// mode 0 (FWD): out = tanh(acc + bias[n])                  -> outB
// mode 1 (ERR): out = RP[m,n] - tanh(acc)                  -> outB (+ 0.5*sum e^2)
// mode 2 (UPD): out = RP[m,n] + LR*(acc - E[m,n])          -> outB (+ optional outF)
//
// Block: 256 threads = 8 waves; wave tile 64x64; block tile 128x256; K step 32.
// WMMA lane layouts (CDNA5 ISA 7.12.2, 16-bit):
//   A 16x32 : lane m=l&15, half=l>>4; elems 0..7 -> K = k0+half*8+j
//                                     elems 8..15 -> K = k0+16+half*8+(j-8)
//   B 32x16 : lane n=l&15, half=l>>4; same K mapping as A  (from BT rows)
//   C/D     : lane n=l&15; vgpr v -> m = half*8 + v
// ---------------------------------------------------------------------------
__global__ __launch_bounds__(256) void pc_gemm_bf16_wmma(
    const __bf16* __restrict__ A, const __bf16* __restrict__ BT,
    int N, int K,
    const __bf16* __restrict__ RP, const __bf16* __restrict__ E,
    const float* __restrict__ bias,
    __bf16* __restrict__ outB, float* __restrict__ outF,
    float* __restrict__ sqDst, int mode, int doSq)
{
    const int tid   = threadIdx.x;
    const int wave  = tid >> 5;
    const int lane  = tid & 31;
    const int waveM = wave >> 2;          // 0..1
    const int waveN = wave & 3;           // 0..3
    const int half  = lane >> 4;          // 0..1
    const int l16   = lane & 15;
    const int m0 = blockIdx.y * 128 + waveM * 64;
    const int n0 = blockIdx.x * 256 + waveN * 64;

    v8f acc[4][4];
    const v8f vzero = {0.f, 0.f, 0.f, 0.f, 0.f, 0.f, 0.f, 0.f};
#pragma unroll
    for (int mi = 0; mi < 4; ++mi)
#pragma unroll
        for (int ni = 0; ni < 4; ++ni) acc[mi][ni] = vzero;

    for (int k0 = 0; k0 < K; k0 += 32) {
        FragAB a[4], b[4];
        const __bf16* Abase = A + (size_t)(m0 + l16) * K + k0;
#pragma unroll
        for (int mi = 0; mi < 4; ++mi) {
            const __bf16* p = Abase + (size_t)mi * 16 * K;
            a[mi].q[0] = *(const u32x4*)(p + half * 8);        // K = k0 + half*8 ..+7
            a[mi].q[1] = *(const u32x4*)(p + 16 + half * 8);   // K = k0+16+half*8 ..+7
        }
        const __bf16* Bbase = BT + (size_t)(n0 + l16) * K + k0;
#pragma unroll
        for (int ni = 0; ni < 4; ++ni) {
            const __bf16* p = Bbase + (size_t)ni * 16 * K;
            b[ni].q[0] = *(const u32x4*)(p + half * 8);        // K = k0 + half*8 ..+7
            b[ni].q[1] = *(const u32x4*)(p + 16 + half * 8);   // K = k0+16+half*8 ..+7
        }
#pragma unroll
        for (int mi = 0; mi < 4; ++mi)
#pragma unroll
            for (int ni = 0; ni < 4; ++ni)
                acc[mi][ni] = __builtin_amdgcn_wmma_f32_16x16x32_bf16(
                    false, a[mi].v, false, b[ni].v,
                    (short)0, acc[mi][ni], false, false);
    }

    float localSq = 0.0f;
#pragma unroll
    for (int mi = 0; mi < 4; ++mi) {
#pragma unroll
        for (int ni = 0; ni < 4; ++ni) {
            const int gmBase = m0 + mi * 16 + half * 8;
            const int gn     = n0 + ni * 16 + l16;
#pragma unroll
            for (int v = 0; v < 8; ++v) {
                const size_t idx = (size_t)(gmBase + v) * N + gn;
                const float a = acc[mi][ni][v];
                float val;
                if (mode == 0) {
                    val = fast_tanh(a + bias[gn]);
                } else if (mode == 1) {
                    val = bf2f(RP[idx]) - fast_tanh(a);
                    if (doSq) localSq += val * val;
                } else {
                    val = bf2f(RP[idx]) + 0.1f * (a - bf2f(E[idx]));
                    if (outF) outF[idx] = val;
                }
                outB[idx] = f2bf(val);
            }
        }
    }

    if (doSq) {
        __shared__ float red[256];
        red[tid] = localSq;
        __syncthreads();
        for (int s = 128; s > 0; s >>= 1) {
            if (tid < s) red[tid] += red[tid + s];
            __syncthreads();
        }
        if (tid == 0) atomicAdd(sqDst, 0.5f * red[0]);
    }
}

// ---------------------------------------------------------------------------
// Helpers: f32 -> bf16 convert, tiled transpose-convert, reductions.
// ---------------------------------------------------------------------------
__global__ void pc_convert_bf16(const float* __restrict__ s, __bf16* __restrict__ d, int n) {
    for (int i = blockIdx.x * blockDim.x + threadIdx.x; i < n; i += gridDim.x * blockDim.x)
        d[i] = f2bf(s[i]);
}

// src is R x C (row-major f32); dst is C x R (row-major bf16) = src^T.
__global__ void pc_transpose_bf16(const float* __restrict__ src, __bf16* __restrict__ dst,
                                  int R, int C) {
    __shared__ float t[32][33];
    const int bc = blockIdx.x * 32;
    const int br = blockIdx.y * 32;
    for (int j = threadIdx.y; j < 32; j += 8)
        t[j][threadIdx.x] = src[(size_t)(br + j) * C + (bc + threadIdx.x)];
    __syncthreads();
    for (int j = threadIdx.y; j < 32; j += 8)
        dst[(size_t)(bc + j) * R + (br + threadIdx.x)] = f2bf(t[threadIdx.x][j]);
}

__global__ void pc_zero_scalar(float* p) {
    if (threadIdx.x == 0 && blockIdx.x == 0) *p = 0.0f;
}

// dst += 0.5 * sum(v[i]^2)
__global__ void pc_sumsq_f32(const float* __restrict__ v, int n, float* dst) {
    __shared__ float red[256];
    float s = 0.0f;
    for (int i = blockIdx.x * blockDim.x + threadIdx.x; i < n; i += gridDim.x * blockDim.x) {
        float x = v[i];
        s += x * x;
    }
    red[threadIdx.x] = s;
    __syncthreads();
    for (int st = 128; st > 0; st >>= 1) {
        if (threadIdx.x < st) red[threadIdx.x] += red[threadIdx.x + st];
        __syncthreads();
    }
    if (threadIdx.x == 0) atomicAdd(dst, 0.5f * red[0]);
}

// ---------------------------------------------------------------------------
// Host-side orchestration.
// ---------------------------------------------------------------------------
extern "C" void kernel_launch(void* const* d_in, const int* in_sizes, int n_in,
                              void* d_out, int out_size, void* d_ws, size_t ws_size,
                              hipStream_t stream) {
    constexpr int B  = 4096;
    constexpr int D0 = 1024, D1 = 2048, D2 = 2048, D3 = 512;
    constexpr int STEPS = 20;

    const float* x  = (const float*)d_in[0];
    const float* W0 = (const float*)d_in[1];
    const float* b0 = (const float*)d_in[2];
    const float* W1 = (const float*)d_in[3];
    const float* b1 = (const float*)d_in[4];
    const float* W2 = (const float*)d_in[5];
    const float* b2 = (const float*)d_in[6];

    float* out  = (float*)d_out;
    float* errp = out + (size_t)B * D3;     // scalar total_error slot

    // Workspace carve-out (bf16 mirrors), 256B aligned.
    char* w = (char*)d_ws;
    size_t off = 0;
    auto alloc = [&](size_t elems) -> __bf16* {
        void* p = w + off;
        off = (off + elems * sizeof(__bf16) + 255) & ~(size_t)255;
        return (__bf16*)p;
    };
    __bf16* xb  = alloc((size_t)B * D0);
    __bf16* Wb0 = alloc((size_t)D1 * D0);   // W0 as-is   (used as BT for @W0^T)
    __bf16* Wb1 = alloc((size_t)D2 * D1);
    __bf16* Wb2 = alloc((size_t)D3 * D2);
    __bf16* Wt0 = alloc((size_t)D0 * D1);   // W0^T       (used as BT for @W0)
    __bf16* Wt1 = alloc((size_t)D1 * D2);
    __bf16* Wt2 = alloc((size_t)D2 * D3);
    __bf16* rb1 = alloc((size_t)B * D1);
    __bf16* rb2 = alloc((size_t)B * D2);
    __bf16* rb3 = alloc((size_t)B * D3);
    __bf16* eb0 = alloc((size_t)B * D0);
    __bf16* eb1 = alloc((size_t)B * D1);
    __bf16* eb2 = alloc((size_t)B * D2);
    if (off > ws_size) return;  // workspace too small; nothing sane to do

    pc_zero_scalar<<<1, 1, 0, stream>>>(errp);

    // bf16 conversions
    auto conv = [&](const float* s, __bf16* d, int n) {
        int blocks = (n + 255) / 256; if (blocks > 4096) blocks = 4096;
        pc_convert_bf16<<<blocks, 256, 0, stream>>>(s, d, n);
    };
    conv(x,  xb,  B * D0);
    conv(W0, Wb0, D1 * D0);
    conv(W1, Wb1, D2 * D1);
    conv(W2, Wb2, D3 * D2);
    pc_transpose_bf16<<<dim3(D0 / 32, D1 / 32), dim3(32, 8), 0, stream>>>(W0, Wt0, D1, D0);
    pc_transpose_bf16<<<dim3(D1 / 32, D2 / 32), dim3(32, 8), 0, stream>>>(W1, Wt1, D2, D1);
    pc_transpose_bf16<<<dim3(D2 / 32, D3 / 32), dim3(32, 8), 0, stream>>>(W2, Wt2, D3, D2);

    auto gemm = [&](const __bf16* A, const __bf16* BT, int N, int K,
                    const __bf16* RP, const __bf16* E, const float* bias,
                    __bf16* outB, float* outF, float* sq, int mode, int doSq) {
        dim3 grid(N / 256, B / 128);
        pc_gemm_bf16_wmma<<<grid, 256, 0, stream>>>(A, BT, N, K, RP, E, bias,
                                                    outB, outF, sq, mode, doSq);
    };

    // Feedforward init: r[i+1] = tanh(r[i] @ W_i^T + b_i)   (BT = W_i)
    gemm(xb,  Wb0, D1, D0, nullptr, nullptr, b0, rb1, nullptr, nullptr, 0, 0);
    gemm(rb1, Wb1, D2, D1, nullptr, nullptr, b1, rb2, nullptr, nullptr, 0, 0);
    gemm(rb2, Wb2, D3, D2, nullptr, nullptr, b2, rb3, nullptr, nullptr, 0, 0);

    // Relaxation steps
    for (int s = 0; s < STEPS; ++s) {
        // errs[i] = r[i] - tanh(r[i+1] @ W_i)   (BT = W_i^T)
        gemm(rb3, Wt2, D2, D3, rb2, nullptr, nullptr, eb2, nullptr, nullptr, 1, 0);
        gemm(rb2, Wt1, D1, D2, rb1, nullptr, nullptr, eb1, nullptr, nullptr, 1, 0);
        gemm(rb1, Wt0, D0, D1, xb,  nullptr, nullptr, eb0, nullptr, nullptr, 1, 0);
        // r[i] += LR*(-errs[i] + errs[i-1] @ W_{i-1}^T)   (BT = W_{i-1})
        gemm(eb0, Wb0, D1, D0, rb1, eb1, nullptr, rb1, nullptr, nullptr, 2, 0);
        gemm(eb1, Wb1, D2, D1, rb2, eb2, nullptr, rb2, nullptr, nullptr, 2, 0);
        gemm(eb2, Wb2, D3, D2, rb3, rb3, nullptr, rb3, out,     nullptr, 2, 0);  // errs[3]=r3
    }

    // Final errors + 0.5*sum(e^2); errs[3] = r3 contributes 0.5*||r3||^2.
    gemm(rb3, Wt2, D2, D3, rb2, nullptr, nullptr, eb2, nullptr, errp, 1, 1);
    gemm(rb2, Wt1, D1, D2, rb1, nullptr, nullptr, eb1, nullptr, errp, 1, 1);
    gemm(rb1, Wt0, D0, D1, xb,  nullptr, nullptr, eb0, nullptr, errp, 1, 1);
    pc_sumsq_f32<<<512, 256, 0, stream>>>(out, B * D3, errp);
}
